// SimpleAttention_39926015984073
// MI455X (gfx1250) — matrix operand order repaired and checked by reading, not compiled, not run on hardware
//
#include <hip/hip_runtime.h>
#include <hip/hip_bf16.h>
#include <math.h>

// ---------------- types / constants ----------------
typedef __attribute__((ext_vector_type(16))) __bf16   v16bf;
typedef __attribute__((ext_vector_type(8)))  float    v8f;
typedef __attribute__((ext_vector_type(8)))  unsigned v8u;

#define B_  4
#define S_  2048
#define E_  256
#define H_  512
#define INV_SCALE 0.04419417382415922f   // 1/sqrt(512)

// ---------------- helpers ----------------
static __device__ __forceinline__ unsigned pack_bf2(float x, float y) {
    __bf16 a = (__bf16)x, b = (__bf16)y;
    unsigned short au = __builtin_bit_cast(unsigned short, a);
    unsigned short bu = __builtin_bit_cast(unsigned short, b);
    return (unsigned)au | ((unsigned)bu << 16);
}

// A fragment, 16x32 (MxK), bf16 row-major source.
// Per-lane data = two contiguous 16B chunks at byte offsets half*16 and half*16+32.
static __device__ __forceinline__ v16bf loadA_bf(const __bf16* base, int ld, int lane) {
    int half = lane >> 4, m = lane & 15;
    const char* p = (const char*)(base + m * ld) + half * 16;
    uint4 lo = *(const uint4*)(p);
    uint4 hi = *(const uint4*)(p + 32);
    v8u u = {lo.x, lo.y, lo.z, lo.w, hi.x, hi.y, hi.z, hi.w};
    return __builtin_bit_cast(v16bf, u);
}

// B fragment, 32x16 (KxN), source column-major: element(k,n)=base[n*ld+k].
// Per-lane data = two contiguous 16B chunks at byte offsets half*16 and half*16+32.
static __device__ __forceinline__ v16bf loadB_bf_col(const __bf16* base, int ld, int lane) {
    int half = lane >> 4, n = lane & 15;
    const char* p = (const char*)(base + n * ld) + half * 16;
    uint4 lo = *(const uint4*)(p);
    uint4 hi = *(const uint4*)(p + 32);
    v8u u = {lo.x, lo.y, lo.z, lo.w, hi.x, hi.y, hi.z, hi.w};
    return __builtin_bit_cast(v16bf, u);
}

static __device__ __forceinline__ v8f wmma_bf16(v16bf a, v16bf b, v8f c) {
    return __builtin_amdgcn_wmma_f32_16x16x32_bf16(false, a, false, b, (short)0, c,
                                                   false, false);
}

static __device__ __forceinline__ float wave_sum(float v) {
#pragma unroll
    for (int m = 16; m > 0; m >>= 1) v += __shfl_xor(v, m, 32);
    return v;
}
static __device__ __forceinline__ float wave_max(float v) {
#pragma unroll
    for (int m = 16; m > 0; m >>= 1) v = fmaxf(v, __shfl_xor(v, m, 32));
    return v;
}

// ---------------- kernel 0: f32 -> bf16 bulk cast (one-time) ----------------
__global__ void k_cast(const float* __restrict__ in, __bf16* __restrict__ out, int n) {
    int i = (blockIdx.x * blockDim.x + threadIdx.x) * 8;
    if (i >= n) return;
    float4 a = *(const float4*)(in + i);
    float4 b = *(const float4*)(in + i + 4);
    uint4 u = {pack_bf2(a.x, a.y), pack_bf2(a.z, a.w),
               pack_bf2(b.x, b.y), pack_bf2(b.z, b.w)};
    *(uint4*)(out + i) = u;
}

// ---------------- kernel 1: qk & v projections (2x2 tiles/wave, dual output) ----
// qk[m,n] = sum_k x[m,k]*qkw[n,k] + qkb[n]   (M=8192, N=512, K=256), same for v.
// Also writes v transposed per batch: vT[(b*H + n)*S + s].
__global__ void k_proj(const __bf16* __restrict__ xB,
                       const __bf16* __restrict__ qkwB, const float* __restrict__ qkb,
                       const __bf16* __restrict__ vwB,  const float* __restrict__ vb,
                       __bf16* __restrict__ qkB, __bf16* __restrict__ vT) {
    int wid  = (blockIdx.x * blockDim.x + threadIdx.x) >> 5;
    int lane = threadIdx.x & 31;
    int gm = wid >> 4;   // 0..255  (rows of 32)
    int gn = wid & 15;   // 0..15   (cols of 32)
    int half = lane >> 4, nm = lane & 15;

    const __bf16* A0  = xB   + (size_t)(gm * 32) * E_;
    const __bf16* Bq0 = qkwB + (gn * 32) * E_;
    const __bf16* Bv0 = vwB  + (gn * 32) * E_;

    v8f cq[2][2] = {};
    v8f cv[2][2] = {};
#pragma unroll
    for (int k0 = 0; k0 < E_; k0 += 32) {
        v16bf a[2], bq[2], bv[2];
#pragma unroll
        for (int i = 0; i < 2; ++i) a[i] = loadA_bf(A0 + i * 16 * E_ + k0, E_, lane);
#pragma unroll
        for (int j = 0; j < 2; ++j) {
            bq[j] = loadB_bf_col(Bq0 + j * 16 * E_ + k0, E_, lane);
            bv[j] = loadB_bf_col(Bv0 + j * 16 * E_ + k0, E_, lane);
        }
#pragma unroll
        for (int i = 0; i < 2; ++i)
#pragma unroll
            for (int j = 0; j < 2; ++j) {
                cq[i][j] = wmma_bf16(a[i], bq[j], cq[i][j]);
                cv[i][j] = wmma_bf16(a[i], bv[j], cv[i][j]);
            }
    }
#pragma unroll
    for (int i = 0; i < 2; ++i)
#pragma unroll
        for (int j = 0; j < 2; ++j) {
            int col = gn * 32 + j * 16 + nm;
            float biasq = qkb[col], biasv = vb[col];
#pragma unroll
            for (int r = 0; r < 8; ++r) {
                int row = gm * 32 + i * 16 + half * 8 + r;
                float vv = cv[i][j][r] + biasv;
                qkB[(size_t)row * H_ + col] = (__bf16)(cq[i][j][r] + biasq);
                int b = row >> 11, s = row & 2047;
                vT[((size_t)b * H_ + col) * S_ + s] = (__bf16)vv;
            }
        }
}

// ---------------- kernel 2: aw = qk . qk^T / sqrt(H)  (4x4 tiles/wave) --------
__global__ void k_scores(const __bf16* __restrict__ qkB, float* __restrict__ aw) {
    int wid  = (blockIdx.x * blockDim.x + threadIdx.x) >> 5;
    int lane = threadIdx.x & 31;
    int b  = wid >> 10;         // 0..3
    int t  = wid & 1023;
    int gm = t >> 5, gn = t & 31;   // 32 x 32 groups of 64x64
    int half = lane >> 4, nm = lane & 15;

    const __bf16* A0 = qkB + ((size_t)b * S_ + gm * 64) * H_;
    const __bf16* B0 = qkB + ((size_t)b * S_ + gn * 64) * H_;
    v8f c[4][4] = {};
    for (int k0 = 0; k0 < H_; k0 += 32) {
        v16bf a[4], bf[4];
#pragma unroll
        for (int i = 0; i < 4; ++i) a[i]  = loadA_bf(A0 + (i * 16) * H_ + k0, H_, lane);
#pragma unroll
        for (int j = 0; j < 4; ++j) bf[j] = loadB_bf_col(B0 + (j * 16) * H_ + k0, H_, lane);
#pragma unroll
        for (int i = 0; i < 4; ++i)
#pragma unroll
            for (int j = 0; j < 4; ++j) c[i][j] = wmma_bf16(a[i], bf[j], c[i][j]);
    }
    float* outp = aw + ((size_t)b * S_ + gm * 64) * S_ + gn * 64;
#pragma unroll
    for (int i = 0; i < 4; ++i)
#pragma unroll
        for (int j = 0; j < 4; ++j)
#pragma unroll
            for (int r = 0; r < 8; ++r)
                outp[(size_t)(i * 16 + half * 8 + r) * S_ + j * 16 + nm] =
                    c[i][j][r] * INV_SCALE;
}

// ---------------- kernel 3a/3b: std over valid [:L,:L] block ----------------
__global__ void k_stats_partial(const float* __restrict__ aw,
                                const int* __restrict__ seqlen,
                                float* __restrict__ part) {
    __shared__ float sh1[8], sh2[8];
    int b = blockIdx.y, c = blockIdx.x;     // 128 chunks x 16 rows
    int L = seqlen[b];
    float s1 = 0.f, s2 = 0.f;
    for (int i = c * 16; i < c * 16 + 16; ++i) {
        if (i >= L) break;
        const float* r = aw + ((size_t)b * S_ + i) * S_;
        for (int j = threadIdx.x; j < L; j += 256) {
            float v = r[j];
            s1 += v; s2 += v * v;
        }
    }
    int lane = threadIdx.x & 31, w = threadIdx.x >> 5;
    s1 = wave_sum(s1); s2 = wave_sum(s2);
    if (lane == 0) { sh1[w] = s1; sh2[w] = s2; }
    __syncthreads();
    if (threadIdx.x == 0) {
        float t1 = 0.f, t2 = 0.f;
        for (int i = 0; i < 8; ++i) { t1 += sh1[i]; t2 += sh2[i]; }
        part[(b * 128 + c) * 2 + 0] = t1;
        part[(b * 128 + c) * 2 + 1] = t2;
    }
}

__global__ void k_stats_final(const float* __restrict__ part,
                              const int* __restrict__ seqlen,
                              float* __restrict__ factor) {
    __shared__ float sh1[4], sh2[4];
    int b = blockIdx.x;
    float s1 = 0.f, s2 = 0.f;
    for (int c = threadIdx.x; c < 128; c += 128) {
        s1 += part[(b * 128 + c) * 2 + 0];
        s2 += part[(b * 128 + c) * 2 + 1];
    }
    int lane = threadIdx.x & 31, w = threadIdx.x >> 5;
    s1 = wave_sum(s1); s2 = wave_sum(s2);
    if (lane == 0) { sh1[w] = s1; sh2[w] = s2; }
    __syncthreads();
    if (threadIdx.x == 0) {
        float t1 = 0.f, t2 = 0.f;
        for (int i = 0; i < 4; ++i) { t1 += sh1[i]; t2 += sh2[i]; }
        float L = (float)seqlen[b];
        float n = L * L;
        float mean = t1 / n;
        float sq = fmaxf(t2 - n * mean * mean, 0.f);
        float sd = sqrtf(sq / fmaxf(n - 1.f, 1.f));
        bool ok = (sd != 0.f) && (sd == sd);
        factor[b] = (seqlen[b] > 1) ? (ok ? sd : 0.f) : 1.f;
    }
}

// ---------------- kernel 4: mask-MLP + padding + softmax ----------------
__global__ void k_softmax(const float* __restrict__ aw, const float* __restrict__ am,
                          const int* __restrict__ seqlen, const float* __restrict__ factor,
                          const float* __restrict__ m1w, const float* __restrict__ m1b,
                          const float* __restrict__ m2w, const float* __restrict__ m2b,
                          float* __restrict__ attn_out, __bf16* __restrict__ attn_bf) {
    __shared__ float buf[8];
    int row = blockIdx.x;
    int b   = row >> 11;
    int L   = seqlen[b];
    float f = factor[b];

    float w1[10], bb1[10], w2[10];
#pragma unroll
    for (int k = 0; k < 10; ++k) { w1[k] = m1w[k]; bb1[k] = m1b[k]; w2[k] = m2w[k]; }
    float b2 = m2b[0];

    const float* awr = aw + (size_t)row * S_;
    const float* amr = am + (size_t)row * S_;

    float lg[8];
    float mx = -INFINITY;
#pragma unroll
    for (int q = 0; q < 8; ++q) {
        int j = threadIdx.x + (q << 8);
        float a = amr[j];
        float pm = b2;
#pragma unroll
        for (int k = 0; k < 10; ++k) pm += fmaxf(fmaf(a, w1[k], bb1[k]), 0.f) * w2[k];
        float base = (j < L) ? awr[j] : -INFINITY;
        float l = base + f * pm;
        lg[q] = l;
        mx = fmaxf(mx, l);
    }
    int lane = threadIdx.x & 31, w = threadIdx.x >> 5;
    mx = wave_max(mx);
    if (lane == 0) buf[w] = mx;
    __syncthreads();
    float MX = buf[0];
    for (int i = 1; i < 8; ++i) MX = fmaxf(MX, buf[i]);
    __syncthreads();

    float e[8], s = 0.f;
#pragma unroll
    for (int q = 0; q < 8; ++q) { e[q] = __expf(lg[q] - MX); s += e[q]; }
    s = wave_sum(s);
    if (lane == 0) buf[w] = s;
    __syncthreads();
    float SUM = 0.f;
    for (int i = 0; i < 8; ++i) SUM += buf[i];
    float inv = 1.f / SUM;

    float*  ao = attn_out + (size_t)row * S_;
    __bf16* ab = attn_bf  + (size_t)row * S_;
#pragma unroll
    for (int q = 0; q < 8; ++q) {
        int j = threadIdx.x + (q << 8);
        float v = e[q] * inv;
        ao[j] = v;
        ab[j] = (__bf16)v;
    }
}

// ---------------- kernel 5: attn @ v   (4x4 tiles/wave, B from vT) ------------
__global__ void k_av(const __bf16* __restrict__ attnB, const __bf16* __restrict__ vT,
                     __bf16* __restrict__ tmpB) {
    int wid  = (blockIdx.x * blockDim.x + threadIdx.x) >> 5;
    int lane = threadIdx.x & 31;
    int b  = wid >> 8;           // 0..3
    int t  = wid & 255;
    int gm = t >> 3, gn = t & 7;     // 32 x 8 groups of 64x64
    int half = lane >> 4, nm = lane & 15;

    const __bf16* A0 = attnB + ((size_t)b * S_ + gm * 64) * S_;
    const __bf16* B0 = vT + ((size_t)b * H_ + gn * 64) * S_;  // B(k,n)=B0[n*S_+k]
    v8f c[4][4] = {};
    for (int k0 = 0; k0 < S_; k0 += 32) {
        __builtin_prefetch(A0 + k0 + 128, 0, 0);   // global_prefetch_b8
        v16bf a[4], bf[4];
#pragma unroll
        for (int i = 0; i < 4; ++i) a[i]  = loadA_bf(A0 + (size_t)(i * 16) * S_ + k0, S_, lane);
#pragma unroll
        for (int j = 0; j < 4; ++j) bf[j] = loadB_bf_col(B0 + (size_t)(j * 16) * S_ + k0, S_, lane);
#pragma unroll
        for (int i = 0; i < 4; ++i)
#pragma unroll
            for (int j = 0; j < 4; ++j) c[i][j] = wmma_bf16(a[i], bf[j], c[i][j]);
    }
    __bf16* outp = tmpB + ((size_t)b * S_ + gm * 64) * H_ + gn * 64;
#pragma unroll
    for (int i = 0; i < 4; ++i)
#pragma unroll
        for (int j = 0; j < 4; ++j)
#pragma unroll
            for (int r = 0; r < 8; ++r)
                outp[(size_t)(i * 16 + half * 8 + r) * H_ + j * 16 + nm] =
                    (__bf16)c[i][j][r];
}

// ---------------- kernel 6: output projection (4x4 tiles/wave) ----------------
__global__ void k_out(const __bf16* __restrict__ tmpB, const __bf16* __restrict__ outwB,
                      const float* __restrict__ outb, float* __restrict__ out) {
    int wid  = (blockIdx.x * blockDim.x + threadIdx.x) >> 5;
    int lane = threadIdx.x & 31;
    int gm = wid >> 2;    // 0..127
    int gn = wid & 3;     // 0..3
    int half = lane >> 4, nm = lane & 15;

    const __bf16* A0 = tmpB + (size_t)(gm * 64) * H_;
    const __bf16* B0 = outwB + (gn * 64) * H_;   // B(k,n)=B0[n*H_+k]
    v8f c[4][4] = {};
    for (int k0 = 0; k0 < H_; k0 += 32) {
        v16bf a[4], bf[4];
#pragma unroll
        for (int i = 0; i < 4; ++i) a[i]  = loadA_bf(A0 + (i * 16) * H_ + k0, H_, lane);
#pragma unroll
        for (int j = 0; j < 4; ++j) bf[j] = loadB_bf_col(B0 + (j * 16) * H_ + k0, H_, lane);
#pragma unroll
        for (int i = 0; i < 4; ++i)
#pragma unroll
            for (int j = 0; j < 4; ++j) c[i][j] = wmma_bf16(a[i], bf[j], c[i][j]);
    }
#pragma unroll
    for (int i = 0; i < 4; ++i)
#pragma unroll
        for (int j = 0; j < 4; ++j) {
            int col = gn * 64 + j * 16 + nm;
            float bias = outb[col];
#pragma unroll
            for (int r = 0; r < 8; ++r)
                out[(size_t)(gm * 64 + i * 16 + half * 8 + r) * E_ + col] =
                    c[i][j][r] + bias;
        }
}

// ---------------- launch ----------------
extern "C" void kernel_launch(void* const* d_in, const int* in_sizes, int n_in,
                              void* d_out, int out_size, void* d_ws, size_t ws_size,
                              hipStream_t stream) {
    const float* x    = (const float*)d_in[0];
    // d_in[1] = padding_mask (bool) — unused; derived from seq_lengths
    const float* am   = (const float*)d_in[2];
    const int*   slen = (const int*)  d_in[3];
    const float* qkw  = (const float*)d_in[4];
    const float* qkb  = (const float*)d_in[5];
    const float* vw   = (const float*)d_in[6];
    const float* vb   = (const float*)d_in[7];
    const float* outw = (const float*)d_in[8];
    const float* outb = (const float*)d_in[9];
    const float* m1w  = (const float*)d_in[10];
    const float* m1b  = (const float*)d_in[11];
    const float* m2w  = (const float*)d_in[12];
    const float* m2b  = (const float*)d_in[13];

    float* out_f  = (float*)d_out;                       // (B,S,E)
    float* attn_f = out_f + (size_t)B_ * S_ * E_;        // (B,S,S)

    char* w = (char*)d_ws;
    __bf16* qkB   = (__bf16*)(w);                         //  8 MB
    __bf16* vT    = (__bf16*)(w + 8388608);               //  8 MB  (B,H,S)
    float*  aw    = (float*) (w + 16777216);              // 64 MB
    __bf16* attnB = (__bf16*)(w + 83886080);              // 32 MB
    __bf16* tmpB  = (__bf16*)(w + 117440512);             //  8 MB
    __bf16* xB    = (__bf16*)(w + 125829120);             //  4 MB
    __bf16* qkwB  = (__bf16*)(w + 130023424);             // 256 KB
    __bf16* vwB   = (__bf16*)(w + 130285568);             // 256 KB
    __bf16* outwB = (__bf16*)(w + 130547712);             // 256 KB
    float*  part  = (float*) (w + 130809856);             //  4 KB
    float*  fact  = part + 1024;

    const int NX = B_ * S_ * E_;   // 2,097,152
    const int NW = H_ * E_;        //   131,072
    k_cast<<<NX / 8 / 256, 256, 0, stream>>>(x, xB, NX);
    k_cast<<<NW / 8 / 256, 256, 0, stream>>>(qkw, qkwB, NW);
    k_cast<<<NW / 8 / 256, 256, 0, stream>>>(vw, vwB, NW);
    k_cast<<<NW / 8 / 256, 256, 0, stream>>>(outw, outwB, NW);

    k_proj         <<<512, 256, 0, stream>>>(xB, qkwB, qkb, vwB, vb, qkB, vT);
    k_scores       <<<512, 256, 0, stream>>>(qkB, aw);
    k_stats_partial<<<dim3(128, B_), 256, 0, stream>>>(aw, slen, part);
    k_stats_final  <<<B_, 128, 0, stream>>>(part, slen, fact);
    k_softmax      <<<8192, 256, 0, stream>>>(aw, am, slen, fact, m1w, m1b, m2w, m2b,
                                              attn_f, attnB);
    k_av           <<<128, 256, 0, stream>>>(attnB, vT, tmpB);
    k_out          <<<64, 256, 0, stream>>>(tmpB, outwB, outb, out_f);
}
